// MixtralMoE_87866440942289
// MI455X (gfx1250) — compile-verified, matrix operand order checked by reading, not run on hardware
//
#include <hip/hip_runtime.h>
#include <math.h>

#define T_TOK 2048
#define H_DIM 1024
#define F_DIM 2048
#define E_NUM 8

#define ROWS 32          // token rows per FFN block (two 16-row WMMA M-tiles)
#define FC   128         // F-chunk width (8 waves x 16 cols)

typedef __bf16 bf16;
typedef bf16  v16bf __attribute__((ext_vector_type(16)));
typedef float v8f   __attribute__((ext_vector_type(8)));

// ---------------------------------------------------------------------------
// Workspace layout:
//   int   cnt[E_NUM];
//   int   tok[E_NUM][T_TOK];
//   float wt [E_NUM][T_TOK];
// ---------------------------------------------------------------------------

__global__ void zero_kernel(float* __restrict__ out, int* __restrict__ cnt) {
    size_t i = (size_t)blockIdx.x * blockDim.x + threadIdx.x;
    size_t n = (size_t)T_TOK * H_DIM;
    for (size_t k = i; k < n; k += (size_t)gridDim.x * blockDim.x) out[k] = 0.0f;
    if (i < E_NUM) cnt[i] = 0;
}

// One block per token: logits -> softmax -> top-2 -> renorm -> scatter into
// per-expert compacted lists.
__global__ __launch_bounds__(256)
void router_kernel(const float* __restrict__ x, const float* __restrict__ gw,
                   int* __restrict__ cnt, int* __restrict__ tok,
                   float* __restrict__ wt) {
    __shared__ float red[256][E_NUM];
    const int t   = blockIdx.x;
    const int tid = threadIdx.x;

    float acc[E_NUM];
#pragma unroll
    for (int e = 0; e < E_NUM; ++e) acc[e] = 0.0f;

    const float* xrow = x + (size_t)t * H_DIM;
    for (int h = tid; h < H_DIM; h += 256) {
        const float xv = xrow[h];
        const float* g = gw + (size_t)h * E_NUM;
#pragma unroll
        for (int e = 0; e < E_NUM; ++e) acc[e] += xv * g[e];
    }
#pragma unroll
    for (int e = 0; e < E_NUM; ++e) red[tid][e] = acc[e];
    __syncthreads();

    for (int s = 128; s > 0; s >>= 1) {
        if (tid < s) {
#pragma unroll
            for (int e = 0; e < E_NUM; ++e) red[tid][e] += red[tid + s][e];
        }
        __syncthreads();
    }

    if (tid == 0) {
        float l[E_NUM];
#pragma unroll
        for (int e = 0; e < E_NUM; ++e) l[e] = red[0][e];
        float mx = l[0];
#pragma unroll
        for (int e = 1; e < E_NUM; ++e) mx = fmaxf(mx, l[e]);
        float p[E_NUM];
#pragma unroll
        for (int e = 0; e < E_NUM; ++e) p[e] = __expf(l[e] - mx);
        int e0 = 0;
#pragma unroll
        for (int e = 1; e < E_NUM; ++e) if (p[e] > p[e0]) e0 = e;
        int e1 = (e0 == 0) ? 1 : 0;
#pragma unroll
        for (int e = 0; e < E_NUM; ++e) if (e != e0 && p[e] > p[e1]) e1 = e;
        const float s  = p[e0] + p[e1];
        const float rs = __builtin_amdgcn_rcpf(s);
        const float w0 = p[e0] * rs;
        const float w1 = p[e1] * rs;

        int pos0 = atomicAdd(&cnt[e0], 1);
        tok[e0 * T_TOK + pos0] = t;
        wt [e0 * T_TOK + pos0] = w0;
        int pos1 = atomicAdd(&cnt[e1], 1);
        tok[e1 * T_TOK + pos1] = t;
        wt [e1 * T_TOK + pos1] = w1;
    }
}

// K-index of fragment half `hi` (0..15) for this lane's half-selector lh(0|8):
// VGPR g = hi/8 covers K base g*16; halves walk K within the lane's 8-slice.
__device__ __forceinline__ int frag_k(int hi, int lh) {
    return ((hi >> 3) << 4) + lh + (hi & 7);
}

// Fast SiLU: hardware v_rcp_f32 instead of the IEEE division chain; result is
// immediately truncated to bf16, so 1-ulp rcp error is irrelevant.
__device__ __forceinline__ float fast_silu(float g) {
    return g * __builtin_amdgcn_rcpf(1.0f + __expf(-g));
}

__global__ __launch_bounds__(256)
void moe_ffn_kernel(const float* __restrict__ x,
                    const float* __restrict__ w1,
                    const float* __restrict__ w2,
                    const float* __restrict__ w3,
                    const int* __restrict__ cnt,
                    const int* __restrict__ tok,
                    const float* __restrict__ wt,
                    float* __restrict__ out) {
    const int e = blockIdx.x;
    const int n = cnt[e];                 // uniform scalar per block
    const int r0 = blockIdx.y * ROWS;
    if (r0 >= n) return;                  // uniform early-exit (before any WMMA)

    __shared__ bf16 Xs[ROWS][H_DIM];      // 64 KB  (bf16 activations)
    __shared__ bf16 Is[ROWS][FC];         //  8 KB  (SwiGLU intermediate chunk)
    __shared__ int   toks[ROWS];
    __shared__ float wts[ROWS];

    const int tid  = threadIdx.x;
    const int lane = tid & 31;
    const int wave = tid >> 5;            // 0..7
    const int ln   = lane & 15;           // N (or M) index within tile
    const int lh   = (lane >> 4) ? 8 : 0; // K half-slice selector
    const int mh   = (lane >> 4) ? 8 : 0; // C-matrix row offset (M = i + mh)

    if (tid < ROWS) {
        const int r = r0 + tid;
        toks[tid] = (r < n) ? tok[e * T_TOK + r] : -1;
        wts[tid]  = (r < n) ? wt [e * T_TOK + r] : 0.0f;
    }
    __syncthreads();

    // Stage X tile (gathered rows) into LDS as bf16.
    for (int i = tid; i < ROWS * H_DIM; i += 256) {
        const int r = i >> 10;            // / H_DIM
        const int h = i & (H_DIM - 1);
        const int tk = toks[r];
        const float v = (tk >= 0) ? x[(size_t)tk * H_DIM + h] : 0.0f;
        Xs[r][h] = (bf16)v;
    }
    __syncthreads();

    const float* w1e = w1 + (size_t)e * H_DIM * F_DIM;
    const float* w3e = w3 + (size_t)e * H_DIM * F_DIM;
    const float* w2e = w2 + (size_t)e * F_DIM * H_DIM;

    // Output accumulators: 2 M-tiles x 8 H-column tiles, fp32.
    v8f oacc[2][8];
#pragma unroll
    for (int mt = 0; mt < 2; ++mt)
#pragma unroll
        for (int j = 0; j < 8; ++j)
#pragma unroll
            for (int i = 0; i < 8; ++i) oacc[mt][j][i] = 0.0f;

    for (int fc = 0; fc < F_DIM; fc += FC) {
        // ---- Phase A: inter[ROWS][FC] = silu(X@w1) * (X@w3) for this chunk.
        const int f0 = fc + wave * 16;
        v8f h1a[2], h3a[2];
#pragma unroll
        for (int mt = 0; mt < 2; ++mt)
#pragma unroll
            for (int i = 0; i < 8; ++i) { h1a[mt][i] = 0.0f; h3a[mt][i] = 0.0f; }

        for (int k0 = 0; k0 < H_DIM; k0 += 32) {
            // Prefetch next K-step's strided rows (global_prefetch_b8).
            if (k0 + 32 < H_DIM) {
                const size_t poff = (size_t)(k0 + 32 + lh) * F_DIM + (f0 + ln);
                __builtin_prefetch(&w1e[poff], 0, 0);
                __builtin_prefetch(&w3e[poff], 0, 0);
            }
            v16bf b1, b3;
#pragma unroll
            for (int hi = 0; hi < 16; ++hi) {
                const int K = frag_k(hi, lh);
                const size_t off = (size_t)(k0 + K) * F_DIM + (f0 + ln);
                b1[hi] = (bf16)w1e[off];
                b3[hi] = (bf16)w3e[off];
            }
#pragma unroll
            for (int mt = 0; mt < 2; ++mt) {
                v16bf a;
#pragma unroll
                for (int hi = 0; hi < 16; ++hi)
                    a[hi] = Xs[mt * 16 + ln][k0 + frag_k(hi, lh)];
                h1a[mt] = __builtin_amdgcn_wmma_f32_16x16x32_bf16(
                    false, a, false, b1, (short)0, h1a[mt], false, false);
                h3a[mt] = __builtin_amdgcn_wmma_f32_16x16x32_bf16(
                    false, a, false, b3, (short)0, h3a[mt], false, false);
            }
        }
        // SwiGLU and stage to LDS (bf16).
#pragma unroll
        for (int mt = 0; mt < 2; ++mt)
#pragma unroll
            for (int i = 0; i < 8; ++i) {
                const float v = fast_silu(h1a[mt][i]) * h3a[mt][i];
                Is[mt * 16 + i + mh][wave * 16 + ln] = (bf16)v;
            }
        __syncthreads();

        // ---- Phase B: oacc += inter_chunk @ w2[fc:fc+FC, :]
#pragma unroll
        for (int kk = 0; kk < FC; kk += 32) {
            v16bf a2[2];
#pragma unroll
            for (int mt = 0; mt < 2; ++mt)
#pragma unroll
                for (int hi = 0; hi < 16; ++hi)
                    a2[mt][hi] = Is[mt * 16 + ln][kk + frag_k(hi, lh)];
#pragma unroll
            for (int j = 0; j < 8; ++j) {
                const int hb = (wave * 8 + j) * 16;
                v16bf b2;
#pragma unroll
                for (int hi = 0; hi < 16; ++hi) {
                    const int K = frag_k(hi, lh);
                    b2[hi] = (bf16)w2e[(size_t)(fc + kk + K) * H_DIM + (hb + ln)];
                }
#pragma unroll
                for (int mt = 0; mt < 2; ++mt)
                    oacc[mt][j] = __builtin_amdgcn_wmma_f32_16x16x32_bf16(
                        false, a2[mt], false, b2, (short)0, oacc[mt][j], false, false);
            }
        }
        __syncthreads();
    }

    // ---- Scatter: out[token, :] += combine_w * expert_out row.
#pragma unroll
    for (int mt = 0; mt < 2; ++mt) {
#pragma unroll
        for (int j = 0; j < 8; ++j) {
            const int hb = (wave * 8 + j) * 16;
#pragma unroll
            for (int i = 0; i < 8; ++i) {
                const int row = mt * 16 + i + mh;
                const int tk  = toks[row];
                if (tk >= 0)
                    atomicAdd(&out[(size_t)tk * H_DIM + hb + ln],
                              oacc[mt][j][i] * wts[row]);
            }
        }
    }
}

extern "C" void kernel_launch(void* const* d_in, const int* in_sizes, int n_in,
                              void* d_out, int out_size, void* d_ws, size_t ws_size,
                              hipStream_t stream) {
    (void)in_sizes; (void)n_in; (void)out_size; (void)ws_size;
    const float* x  = (const float*)d_in[0];   // [T, H]
    const float* gw = (const float*)d_in[1];   // [H, E]
    const float* w1 = (const float*)d_in[2];   // [E, H, F]
    const float* w2 = (const float*)d_in[3];   // [E, F, H]
    const float* w3 = (const float*)d_in[4];   // [E, H, F]
    float* out = (float*)d_out;                // [T, H]

    int*   cnt = (int*)d_ws;
    int*   tok = cnt + E_NUM;
    float* wtb = (float*)(tok + E_NUM * T_TOK);

    zero_kernel<<<512, 256, 0, stream>>>(out, cnt);
    router_kernel<<<T_TOK, 256, 0, stream>>>(x, gw, cnt, tok, wtb);

    dim3 grid(E_NUM, T_TOK / ROWS);
    moe_ffn_kernel<<<grid, 256, 0, stream>>>(x, w1, w2, w3, cnt, tok, wtb, out);
}
